// SAE_89833535963398
// MI455X (gfx1250) — matrix operand order repaired and checked by reading, not compiled, not run on hardware
//
#include <hip/hip_runtime.h>

#define B_ROWS  8192
#define D_INP   256
#define HIDDEN  8192
#define K_TOP   32

// LDS A-tile row stride (elements): 256 + 8 pad -> 528B rows, bank-conflict-free
#define A_LDS_STRIDE 264

typedef __attribute__((ext_vector_type(16))) __bf16 v16bf;
typedef __attribute__((ext_vector_type(8)))  __bf16 v8bf;
typedef __attribute__((ext_vector_type(8)))  float  v8f;

static_assert(sizeof(v16bf) == 32, "v16bf must be 32 bytes");

// ---------------------------------------------------------------------------
// d_out layout (floats, reference tuple order):
//   recon  [8192*256]  @ 0
//   fvu    [1]         @ 2097152
//   z      [8192*8192] @ 2097153
//   topkid [8192*32]   @ 69206017   (written as float(index))
// ---------------------------------------------------------------------------

// ws layout (bytes):
//   [ 0MB,  4MB)  Xb   bf16  (x - b_dec)
//   [ 4MB,  8MB)  Wb   bf16  (W_enc)
//   [ 8MB,  9MB)  idxi int32 topk indices
//   [ 9MB, 10MB)  vals f32   topk values
//   [10MB, ... )  accum[0]=sse, accum[1]=tv

__global__ void sae_init(float* __restrict__ acc) {
  acc[0] = 0.0f;
  acc[1] = 0.0f;
}

// Convert x - b_dec and W_enc to bf16 (both are 8192*256 elements).
__global__ __launch_bounds__(256) void sae_convert(
    const float* __restrict__ x, const float* __restrict__ b_dec,
    const float* __restrict__ W_enc,
    __bf16* __restrict__ Xb, __bf16* __restrict__ Wb) {
  size_t i = (size_t)blockIdx.x * blockDim.x + threadIdx.x;
  if (i < (size_t)B_ROWS * D_INP) {
    Xb[i] = (__bf16)(x[i] - b_dec[i & (D_INP - 1)]);
    Wb[i] = (__bf16)(W_enc[i]);
  }
}

// Encoder GEMM: acts = relu((x-b_dec) @ W_enc^T + b_enc), written into zbuf.
// Block = 16 rows x 512 cols. The shared 16x256 A tile is staged into LDS
// once per block with GLOBAL_LOAD_ASYNC_TO_LDS_B128 (ASYNCcnt), then each of
// the 8 waves computes a 16x64 strip: 4 f32 accumulators, K-loop of 8 steps
// of v_wmma_f32_16x16x32_bf16.
__global__ __launch_bounds__(256) void sae_encode_wmma(
    const __bf16* __restrict__ Xb, const __bf16* __restrict__ Wb,
    const float* __restrict__ b_enc, float* __restrict__ zbuf) {
  __shared__ __bf16 sA[16 * A_LDS_STRIDE];   // 8.25 KB padded A tile

  const int tid  = threadIdx.x;
  const int lane = tid & 31;
  const int wIdx = tid >> 5;                 // wave in block: 0..7
  const int nb   = blockIdx.x & 15;          // 16 column blocks of 512
  const int mb   = blockIdx.x >> 4;          // 512 row tiles of 16
  const int M0   = mb * 16;
  const int N0   = nb * 512 + wIdx * 64;
  const int l15  = lane & 15;
  const int hi   = lane >> 4;                // 0: lanes 0-15, 1: lanes 16-31

  // ---- async stage A tile (16 rows x 512B) into LDS: 512 x 16B transfers,
  //      2 per thread. LDS rows padded to 528B for conflict-free ds reads.
  {
    const unsigned sA_base = (unsigned)(uintptr_t)(void*)sA;  // LDS offset
#pragma unroll
    for (int rep = 0; rep < 2; ++rep) {
      const int id  = tid + rep * 256;       // 0..511
      const int row = id >> 5;               // 0..15
      const int seg = id & 31;               // 16B segment within 512B row
      const __bf16* gsrc = Xb + (size_t)(M0 + row) * D_INP + seg * 8;
      const unsigned ldst = sA_base + (unsigned)(row * A_LDS_STRIDE + seg * 8) * 2u;
      asm volatile("global_load_async_to_lds_b128 %0, %1, off"
                   :: "v"(ldst), "v"(gsrc) : "memory");
    }
    asm volatile("s_wait_asynccnt 0x0" ::: "memory");
  }
  __syncthreads();

  // B fragment source: 16 contiguous bf16 of W_enc row (N + l15), K base hi*16.
  const __bf16* w0 = Wb + (size_t)(N0 +  0 + l15) * D_INP + hi * 16;
  const __bf16* w1 = Wb + (size_t)(N0 + 16 + l15) * D_INP + hi * 16;
  const __bf16* w2 = Wb + (size_t)(N0 + 32 + l15) * D_INP + hi * 16;
  const __bf16* w3 = Wb + (size_t)(N0 + 48 + l15) * D_INP + hi * 16;

  // A fragment source (LDS): row M0+l15; K offset per lane-half is hi*16
  // (16 contiguous K, matching the B fragment order).
  const __bf16* arow = sA + l15 * A_LDS_STRIDE;

  v8f c0 = {}, c1 = {}, c2 = {}, c3 = {};

#pragma unroll
  for (int k0 = 0; k0 < D_INP; k0 += 32) {
    v8bf alo = *(const v8bf*)(arow + k0 + hi * 16);
    v8bf ahi = *(const v8bf*)(arow + k0 + hi * 16 + 8);
    v16bf a = __builtin_shufflevector(alo, ahi,
        0, 1, 2, 3, 4, 5, 6, 7, 8, 9, 10, 11, 12, 13, 14, 15);
    v16bf b0 = *(const v16bf*)(w0 + k0);
    v16bf b1 = *(const v16bf*)(w1 + k0);
    v16bf b2 = *(const v16bf*)(w2 + k0);
    v16bf b3 = *(const v16bf*)(w3 + k0);
    c0 = __builtin_amdgcn_wmma_f32_16x16x32_bf16(false, a, false, b0, (short)0, c0, false, false);
    c1 = __builtin_amdgcn_wmma_f32_16x16x32_bf16(false, a, false, b1, (short)0, c1, false, false);
    c2 = __builtin_amdgcn_wmma_f32_16x16x32_bf16(false, a, false, b2, (short)0, c2, false, false);
    c3 = __builtin_amdgcn_wmma_f32_16x16x32_bf16(false, a, false, b3, (short)0, c3, false, false);
  }

  const int col = N0 + l15;
  const float be0 = b_enc[col +  0];
  const float be1 = b_enc[col + 16];
  const float be2 = b_enc[col + 32];
  const float be3 = b_enc[col + 48];
#pragma unroll
  for (int v = 0; v < 8; ++v) {
    // C/D layout: VGPR v holds M=v (lanes 0-15) / M=v+8 (lanes 16-31), N=l15.
    float* zr = zbuf + (size_t)(M0 + v + hi * 8) * HIDDEN + col;
    zr[ 0] = fmaxf(c0[v] + be0, 0.0f);
    zr[16] = fmaxf(c1[v] + be1, 0.0f);
    zr[32] = fmaxf(c2[v] + be2, 0.0f);
    zr[48] = fmaxf(c3[v] + be3, 0.0f);
  }
}

// Per-row top-k via 4-pass byte radix select on float bits (acts >= 0 so bit
// pattern order == value order). Entire 32KB row staged in LDS: one global
// read + one global write per element.
__global__ __launch_bounds__(256) void sae_topk(
    float* __restrict__ zbuf, int* __restrict__ idxi,
    float* __restrict__ vals, float* __restrict__ oidx) {
  __shared__ float    lrow[HIDDEN];          // 32 KB of the 320 KB/WGP LDS
  __shared__ unsigned hist[256];
  __shared__ unsigned s_prefix, s_rem, s_tie, s_out;

  const int row = blockIdx.x;
  const int tid = threadIdx.x;
  float* arow = zbuf + (size_t)row * HIDDEN;

  for (int i = tid; i < HIDDEN; i += 256) lrow[i] = arow[i];
  __syncthreads();

  unsigned prefix = 0, rem = K_TOP;
  for (int shift = 24; shift >= 0; shift -= 8) {
    hist[tid] = 0;
    __syncthreads();
    for (int i = tid; i < HIDDEN; i += 256) {
      unsigned u = __float_as_uint(lrow[i]);
      bool sel = (shift == 24) ? true : ((u >> (shift + 8)) == prefix);
      if (sel) atomicAdd(&hist[(u >> shift) & 0xFFu], 1u);
    }
    __syncthreads();
    if (tid == 0) {
      unsigned cum = 0;
      int b = 255;
      for (; b > 0; --b) {
        unsigned c = hist[b];
        if (cum + c >= rem) break;
        cum += c;
      }
      s_prefix = (prefix << 8) | (unsigned)b;
      s_rem = rem - cum;
    }
    __syncthreads();
    prefix = s_prefix;
    rem = s_rem;
    __syncthreads();
  }

  const unsigned T = prefix;                 // bits of k-th largest value
  if (tid == 0) { s_tie = 0; s_out = 0; }
  __syncthreads();

  for (int i = tid; i < HIDDEN; i += 256) {
    float v = lrow[i];
    unsigned u = __float_as_uint(v);
    bool keep = (u > T);
    if (u == T) {
      unsigned t = atomicAdd(&s_tie, 1u);
      keep = (t < rem);
    }
    if (keep) {
      unsigned p = atomicAdd(&s_out, 1u);
      if (p < K_TOP) {
        idxi[row * K_TOP + p] = i;
        vals[row * K_TOP + p] = v;
        oidx[row * K_TOP + p] = (float)i;
      }
      arow[i] = v;
    } else {
      arow[i] = 0.0f;
    }
  }
}

// Sparse decode + SSE accumulation: recon = z_sparse @ W_dec + b_dec.
__global__ __launch_bounds__(256) void sae_decode(
    const int* __restrict__ idxi, const float* __restrict__ vals,
    const float* __restrict__ W_dec, const float* __restrict__ b_dec,
    const float* __restrict__ x, float* __restrict__ recon,
    float* __restrict__ sse) {
  __shared__ int   sidx[K_TOP];
  __shared__ float sval[K_TOP];
  __shared__ float red[256];
  const int row = blockIdx.x;
  const int tid = threadIdx.x;

  if (tid < K_TOP) {
    sidx[tid] = idxi[row * K_TOP + tid];
    sval[tid] = vals[row * K_TOP + tid];
  }
  __syncthreads();

  float acc = b_dec[tid];
#pragma unroll
  for (int j = 0; j < K_TOP; ++j)
    acc = fmaf(sval[j], W_dec[(size_t)sidx[j] * D_INP + tid], acc);

  recon[(size_t)row * D_INP + tid] = acc;
  float e = acc - x[(size_t)row * D_INP + tid];
  red[tid] = e * e;
  __syncthreads();
  for (int s = 128; s > 0; s >>= 1) {
    if (tid < s) red[tid] += red[tid + s];
    __syncthreads();
  }
  if (tid == 0) atomicAdd(sse, red[0]);
}

// total_variance = sum over all elements of (x - colmean(x))^2. One block,
// thread tid owns column tid -> fully coalesced row-major sweeps.
__global__ __launch_bounds__(256) void sae_variance(
    const float* __restrict__ x, float* __restrict__ tv) {
  __shared__ float red[256];
  const int tid = threadIdx.x;
  float s = 0.0f;
  for (int r = 0; r < B_ROWS; ++r) s += x[(size_t)r * D_INP + tid];
  const float mean = s * (1.0f / (float)B_ROWS);
  float s2 = 0.0f;
  for (int r = 0; r < B_ROWS; ++r) {
    float d = x[(size_t)r * D_INP + tid] - mean;
    s2 = fmaf(d, d, s2);
  }
  red[tid] = s2;
  __syncthreads();
  for (int s_ = 128; s_ > 0; s_ >>= 1) {
    if (tid < s_) red[tid] += red[tid + s_];
    __syncthreads();
  }
  if (tid == 0) *tv = red[0];
}

__global__ void sae_fvu(const float* __restrict__ acc, float* __restrict__ fvu) {
  *fvu = acc[0] / (acc[1] + 1e-8f);
}

extern "C" void kernel_launch(void* const* d_in, const int* in_sizes, int n_in,
                              void* d_out, int out_size, void* d_ws, size_t ws_size,
                              hipStream_t stream) {
  const float* x     = (const float*)d_in[0];
  const float* W_enc = (const float*)d_in[1];
  const float* b_enc = (const float*)d_in[2];
  const float* W_dec = (const float*)d_in[3];
  const float* b_dec = (const float*)d_in[4];
  // d_in[5] == k (compile-time K_TOP = 32)

  float* out = (float*)d_out;
  const size_t OFF_RECON = 0;
  const size_t OFF_FVU   = (size_t)B_ROWS * D_INP;          // 2097152
  const size_t OFF_Z     = OFF_FVU + 1;                     // 2097153
  const size_t OFF_IDX   = OFF_Z + (size_t)B_ROWS * HIDDEN; // 69206017

  float* recon = out + OFF_RECON;
  float* fvu   = out + OFF_FVU;
  float* zbuf  = out + OFF_Z;   // acts written here, then sparsified in place
  float* oidx  = out + OFF_IDX;

  char* ws = (char*)d_ws;
  __bf16* Xb    = (__bf16*)(ws);
  __bf16* Wb    = (__bf16*)(ws + ((size_t)4 << 20));
  int*    idxi  = (int*)   (ws + ((size_t)8 << 20));
  float*  vals  = (float*) (ws + ((size_t)9 << 20));
  float*  accum = (float*) (ws + ((size_t)10 << 20)); // [0]=sse, [1]=tv

  sae_init<<<1, 1, 0, stream>>>(accum);

  const int nconv = (B_ROWS * D_INP) / 256;                 // 8192 blocks
  sae_convert<<<nconv, 256, 0, stream>>>(x, b_dec, W_enc, Xb, Wb);

  // Block covers 16 rows x 512 cols: grid = 512 M-tiles x 16 N-blocks.
  const int blocks = (B_ROWS / 16) * (HIDDEN / 512);        // 8192
  sae_encode_wmma<<<blocks, 256, 0, stream>>>(Xb, Wb, b_enc, zbuf);

  sae_topk<<<B_ROWS, 256, 0, stream>>>(zbuf, idxi, vals, oidx);

  sae_decode<<<B_ROWS, 256, 0, stream>>>(idxi, vals, W_dec, b_dec, x, recon, accum);

  sae_variance<<<1, 256, 0, stream>>>(x, accum + 1);

  sae_fvu<<<1, 1, 0, stream>>>(accum, fvu);
}
